// AttentionBlock_6493990552057
// MI455X (gfx1250) — matrix operand order repaired and checked by reading, not compiled, not run on hardware
//
#include <hip/hip_runtime.h>

typedef __attribute__((ext_vector_type(16))) _Float16 v16h;
typedef __attribute__((ext_vector_type(8)))  _Float16 v8h;
typedef __attribute__((ext_vector_type(8)))  float    v8f;

#define C_DIM 256
#define N_DIM 1024
#define B_DIM 32
#define G_DIM 8
#define EPS_GN 1e-5f

// ---------------------------------------------------------------------------
// WMMA operand loaders (ISA 7.12.2 layouts, 16-bit operands, wave32).
// A (16xK=32): M = lane%16, K split {0-7,16-23}/{8-15,24-31} by lane half.
// src is row-major [M][K] with leading dim ld (in halves).
__device__ __forceinline__ v16h load_opA(const _Float16* base, int ld, int lane) {
    int row  = lane & 15;
    int kofs = (lane >> 4) << 3;                 // 0 or 8
    const _Float16* p = base + row * ld + kofs;
    v8h lo = *(const v8h*)(p);                   // K = kofs .. kofs+7
    v8h hi = *(const v8h*)(p + 16);              // K = kofs+16 .. kofs+23
    return __builtin_shufflevector(lo, hi, 0,1,2,3,4,5,6,7,8,9,10,11,12,13,14,15);
}

// B (K=32 x 16): N = lane%16, K split {0-7,16-23}/{8-15,24-31} by lane half.
// src is [N][K] row-major (i.e. B^T), leading dim ld.
__device__ __forceinline__ v16h load_opB(const _Float16* base, int ld, int lane) {
    const _Float16* p = base + (lane & 15) * ld + ((lane >> 4) << 3);
    v8h lo = *(const v8h*)(p);
    v8h hi = *(const v8h*)(p + 16);
    return __builtin_shufflevector(lo, hi, 0,1,2,3,4,5,6,7,8,9,10,11,12,13,14,15);
}

__device__ __forceinline__ v8f wmma_f16(v16h a, v16h b, v8f c) {
    return __builtin_amdgcn_wmma_f32_16x16x32_f16(false, a, false, b, (short)0, c,
                                                  false, false);
}

// Async DMA of one 16-byte chunk Global -> LDS (CDNA5, tracked by ASYNCcnt).
// Low 32 bits of a generic LDS pointer are the workgroup-relative LDS offset.
__device__ __forceinline__ void async_copy16(const _Float16* gsrc, _Float16* ldst) {
    unsigned loff = (unsigned)(size_t)ldst;
    asm volatile("global_load_async_to_lds_b128 %0, %1, off"
                 :: "v"(loff), "v"(gsrc) : "memory");
}

// ---------------------------------------------------------------------------
__global__ void f32_to_f16_kernel(const float* __restrict__ src,
                                  _Float16* __restrict__ dst, int n) {
    int i = blockIdx.x * blockDim.x + threadIdx.x;
    if (i < n) dst[i] = (_Float16)src[i];
}

// ---------------------------------------------------------------------------
// GroupNorm over (C/G channels x N spatial) per (b,g); writes h as f16 in
// transposed layout [B][N][C] so downstream WMMA operand loads are contiguous.
__global__ void __launch_bounds__(256)
groupnorm_kernel(const float* __restrict__ x, const float* __restrict__ gamma,
                 const float* __restrict__ beta, _Float16* __restrict__ h) {
    const int CG  = C_DIM / G_DIM;               // 32
    const int NPG = CG * N_DIM;                  // 32768
    int b = blockIdx.x / G_DIM;
    int g = blockIdx.x % G_DIM;
    const float* xg = x + ((size_t)b * C_DIM + g * CG) * N_DIM;

    float s = 0.f, s2 = 0.f;
    for (int idx = threadIdx.x; idx < NPG; idx += 256) {
        float v = xg[idx];
        s += v; s2 += v * v;
    }
    #pragma unroll
    for (int off = 16; off > 0; off >>= 1) {
        s  += __shfl_xor(s,  off, 32);
        s2 += __shfl_xor(s2, off, 32);
    }
    __shared__ float rs[8], rs2[8];
    int lane = threadIdx.x & 31, wid = threadIdx.x >> 5;
    if (lane == 0) { rs[wid] = s; rs2[wid] = s2; }
    __syncthreads();
    float ts = 0.f, t2 = 0.f;
    #pragma unroll
    for (int w = 0; w < 8; ++w) { ts += rs[w]; t2 += rs2[w]; }
    float mean = ts * (1.0f / NPG);
    float var  = t2 * (1.0f / NPG) - mean * mean;
    float rstd = rsqrtf(var + EPS_GN);

    for (int idx = threadIdx.x; idx < NPG; idx += 256) {
        int cl = idx / N_DIM;
        int n  = idx - cl * N_DIM;
        int c  = g * CG + cl;
        float v = (xg[idx] - mean) * rstd * gamma[c] + beta[c];
        h[((size_t)b * N_DIM + n) * C_DIM + c] = (_Float16)v;
    }
}

// ---------------------------------------------------------------------------
// 256x256 weight GEMM over h[B][N][C]: out[o][n] = W[o][:] . h[n][:] + bias[o]
// MODE 0: f16 out, transposed store [B][N][C]   (q, k)
// MODE 1: f16 out, direct store     [B][C][N]   (v)
// MODE 2: f32 out = x + D + bias,   [B][C][N]   (final projection + residual)
template <int MODE>
__global__ void __launch_bounds__(256)
gemm256_kernel(const _Float16* __restrict__ W, const float* __restrict__ bias,
               const _Float16* __restrict__ Hin, const float* __restrict__ Xres,
               _Float16* __restrict__ outH, float* __restrict__ outF) {
    int lane = threadIdx.x & 31;
    int wid  = threadIdx.x >> 5;
    int tile = blockIdx.x * 8 + wid;             // 32768 tiles total
    int nt = tile & 63;                          // N/16 = 64
    int ot = (tile >> 6) & 15;                   // C/16 = 16
    int b  = tile >> 10;
    int n0 = nt << 4, o0 = ot << 4;

    const _Float16* hb = Hin + ((size_t)b * N_DIM + n0) * C_DIM;
    const _Float16* wb = W + (size_t)o0 * C_DIM;

    v8f acc = {};
    #pragma unroll
    for (int kc = 0; kc < C_DIM; kc += 32) {
        v16h a  = load_opA(wb + kc, C_DIM, lane);
        v16h bm = load_opB(hb + kc, C_DIM, lane);
        acc = wmma_f16(a, bm, acc);
    }
    int colp = lane & 15;
    int rb   = (lane >> 4) << 3;
    #pragma unroll
    for (int r = 0; r < 8; ++r) {
        int o = o0 + rb + r;
        int n = n0 + colp;
        float v = acc[r] + bias[o];
        if (MODE == 0) {
            outH[((size_t)b * N_DIM + n) * C_DIM + o] = (_Float16)v;
        } else if (MODE == 1) {
            outH[((size_t)b * C_DIM + o) * N_DIM + n] = (_Float16)v;
        } else {
            size_t idx = ((size_t)b * C_DIM + o) * N_DIM + n;
            outF[idx] = Xres[idx] + v;
        }
    }
}

// ---------------------------------------------------------------------------
// Flash attention. Block = 8 waves, one batch; wave owns 16 query rows.
// K/V j-tiles (32 wide) are cooperatively DMA'd Global->LDS with
// global_load_async_to_lds_b128 (double-buffered; ASYNCcnt + one barrier per
// iteration). WMMA B-operands then come from LDS. Scores never touch HBM.
__global__ void __launch_bounds__(256)
flash_attn_kernel(const _Float16* __restrict__ Q,   // [B][N][C]
                  const _Float16* __restrict__ K,   // [B][N][C]
                  const _Float16* __restrict__ V,   // [B][C][N]
                  _Float16* __restrict__ OA) {      // [B][N][C]
    __shared__ _Float16 Kt[2][32 * 256];            // 2 x 16 KB  [j][c]
    __shared__ _Float16 Vt[2][256 * 32];            // 2 x 16 KB  [c][j]
    __shared__ _Float16 Plds[8][16 * 32];           // 8 KB       per-wave P
    int tid  = threadIdx.x;
    int lane = tid & 31;
    int wid  = tid >> 5;
    int b    = blockIdx.x >> 3;
    int i0   = ((((blockIdx.x & 7) << 3) + wid) << 4);
    const float scale = 0.0625f;                    // 1/sqrt(256)

    const _Float16* qb = Q + ((size_t)b * N_DIM + i0) * C_DIM;
    const _Float16* kb = K + (size_t)b * N_DIM * C_DIM;
    const _Float16* vb = V + (size_t)b * C_DIM * N_DIM;
    _Float16* pl = &Plds[wid][0];

    v8f zero = {};
    v8f O[16];
    #pragma unroll
    for (int t = 0; t < 16; ++t) O[t] = zero;
    float mrun[8], lrun[8];
    #pragma unroll
    for (int r = 0; r < 8; ++r) { mrun[r] = -3.0e38f; lrun[r] = 0.f; }

    int colp = lane & 15;
    int rb   = (lane >> 4) << 3;

    // Stage one (K,V) j-tile: 1024 16B chunks each of K (32x256) and V (256x32),
    // 4 chunks per array per thread, all via the async Global->LDS DMA path.
    auto stage = [&](int buf, int j0) {
        #pragma unroll
        for (int p = 0; p < 4; ++p) {
            int e = tid + p * 256;
            async_copy16(kb + (size_t)(j0 + (e >> 5)) * C_DIM + ((e & 31) << 3),
                         &Kt[buf][e << 3]);
            async_copy16(vb + (size_t)(e >> 2) * N_DIM + j0 + ((e & 3) << 3),
                         &Vt[buf][e << 3]);
        }
    };

    stage(0, 0);
    for (int jt = 0; jt < N_DIM / 32; ++jt) {
        int cur = jt & 1;
        asm volatile("s_wait_asynccnt 0" ::: "memory");  // own DMAs for cur done
        __syncthreads();                                  // all waves' DMAs done
        if (jt + 1 < N_DIM / 32) stage(cur ^ 1, (jt + 1) * 32);

        // --- S tiles: rows i0..i0+15, cols j0..j0+31 (K from LDS) ---
        v8f S0 = zero, S1 = zero;
        #pragma unroll
        for (int kc = 0; kc < C_DIM; kc += 32) {
            v16h aq = load_opA(qb + kc, C_DIM, lane);
            v16h b0 = load_opB(&Kt[cur][kc], C_DIM, lane);
            v16h b1 = load_opB(&Kt[cur][16 * C_DIM + kc], C_DIM, lane);
            S0 = wmma_f16(aq, b0, S0);
            S1 = wmma_f16(aq, b1, S1);
        }
        // --- online softmax (per row slot r; rows map to 16-lane halves) ---
        float alpha[8];
        #pragma unroll
        for (int r = 0; r < 8; ++r) {
            float a = S0[r] * scale, c = S1[r] * scale;
            float mx = fmaxf(a, c);
            #pragma unroll
            for (int off = 1; off < 16; off <<= 1)
                mx = fmaxf(mx, __shfl_xor(mx, off, 16));
            float mn = fmaxf(mrun[r], mx);
            alpha[r] = __expf(mrun[r] - mn);
            float e0 = __expf(a - mn);
            float e1 = __expf(c - mn);
            S0[r] = e0; S1[r] = e1;
            float rsum = e0 + e1;
            #pragma unroll
            for (int off = 1; off < 16; off <<= 1)
                rsum += __shfl_xor(rsum, off, 16);
            lrun[r] = lrun[r] * alpha[r] + rsum;
            mrun[r] = mn;
        }
        #pragma unroll
        for (int t = 0; t < 16; ++t)
            #pragma unroll
            for (int r = 0; r < 8; ++r) O[t][r] *= alpha[r];

        // --- transpose P (D layout -> A layout) through per-wave LDS ---
        #pragma unroll
        for (int r = 0; r < 8; ++r) {
            pl[(rb + r) * 32 + colp]      = (_Float16)S0[r];
            pl[(rb + r) * 32 + 16 + colp] = (_Float16)S1[r];
        }
        asm volatile("s_wait_dscnt 0" ::: "memory");  // wave-private RAW on LDS
        v16h aP = load_opA(pl, 32, lane);

        // --- O += P @ V^T over all 16 c-tiles (V from LDS) ---
        #pragma unroll
        for (int t = 0; t < 16; ++t) {
            v16h bv = load_opB(&Vt[cur][(t * 16) * 32], 32, lane);
            O[t] = wmma_f16(aP, bv, O[t]);
        }
    }

    float inv[8];
    #pragma unroll
    for (int r = 0; r < 8; ++r) inv[r] = 1.0f / lrun[r];
    #pragma unroll
    for (int t = 0; t < 16; ++t)
        #pragma unroll
        for (int r = 0; r < 8; ++r) {
            int i = i0 + rb + r;
            int c = t * 16 + colp;
            OA[((size_t)b * N_DIM + i) * C_DIM + c] = (_Float16)(O[t][r] * inv[r]);
        }
}

// ---------------------------------------------------------------------------
extern "C" void kernel_launch(void* const* d_in, const int* in_sizes, int n_in,
                              void* d_out, int out_size, void* d_ws, size_t ws_size,
                              hipStream_t stream) {
    const float* x     = (const float*)d_in[0];
    const float* gamma = (const float*)d_in[1];
    const float* beta  = (const float*)d_in[2];
    const float* wq    = (const float*)d_in[3];
    const float* bq    = (const float*)d_in[4];
    const float* wk    = (const float*)d_in[5];
    const float* bk    = (const float*)d_in[6];
    const float* wv    = (const float*)d_in[7];
    const float* bv    = (const float*)d_in[8];
    const float* wo    = (const float*)d_in[9];
    const float* bo    = (const float*)d_in[10];
    float* out = (float*)d_out;

    const size_t WSZ = (size_t)C_DIM * C_DIM;           // 65536
    const size_t HSZ = (size_t)B_DIM * N_DIM * C_DIM;   // 8.4M halves
    _Float16* wqh = (_Float16*)d_ws;
    _Float16* wkh = wqh + WSZ;
    _Float16* wvh = wkh + WSZ;
    _Float16* woh = wvh + WSZ;
    _Float16* h   = woh + WSZ;
    _Float16* q   = h  + HSZ;
    _Float16* k   = q  + HSZ;
    _Float16* v   = k  + HSZ;
    _Float16* hA  = v  + HSZ;   // total ~84.4 MB of d_ws

    // 1) weights -> f16
    f32_to_f16_kernel<<<256, 256, 0, stream>>>(wq, wqh, (int)WSZ);
    f32_to_f16_kernel<<<256, 256, 0, stream>>>(wk, wkh, (int)WSZ);
    f32_to_f16_kernel<<<256, 256, 0, stream>>>(wv, wvh, (int)WSZ);
    f32_to_f16_kernel<<<256, 256, 0, stream>>>(wo, woh, (int)WSZ);

    // 2) GroupNorm -> h [B][N][C] f16
    groupnorm_kernel<<<B_DIM * G_DIM, 256, 0, stream>>>(x, gamma, beta, h);

    // 3) q/k/v projections (WMMA GEMMs)
    gemm256_kernel<0><<<4096, 256, 0, stream>>>(wqh, bq, h, nullptr, q, nullptr);
    gemm256_kernel<0><<<4096, 256, 0, stream>>>(wkh, bk, h, nullptr, k, nullptr);
    gemm256_kernel<1><<<4096, 256, 0, stream>>>(wvh, bv, h, nullptr, v, nullptr);

    // 4) fused attention (async-DMA staged K/V, scores never hit HBM)
    flash_attn_kernel<<<B_DIM * 8, 256, 0, stream>>>(q, k, v, hA);

    // 5) output projection + bias + residual -> f32 d_out
    gemm256_kernel<2><<<4096, 256, 0, stream>>>(woh, bo, hA, x, nullptr, out);
}
